// _Gate_74655121539085
// MI455X (gfx1250) — matrix operand order repaired and checked by reading, not compiled, not run on hardware
//
#include <hip/hip_runtime.h>

// ---------------------------------------------------------------------------
// MoE gate for MI455X (gfx1250, wave32):
//   Phase 1: logits = x @ W^T via split-bf16 (hi+lo) on V_WMMA_F32_16X16X32_BF16.
//            3 products (hh, hl, lh) give ~2^-16 relative accuracy (fp16+),
//            ~2.7x the fp32-WMMA throughput on the 16-bit matrix pipe.
//   Phase 2: softmax + bias + group-prune + top-8 routing, one wave per token.
// ---------------------------------------------------------------------------

#define GATE_N    8192
#define GATE_DIM  7168
#define GATE_E    256
#define GATE_TOPK 8

typedef float  v8f  __attribute__((ext_vector_type(8)));
typedef __bf16 v8bf __attribute__((ext_vector_type(8)));
typedef __bf16 v16bf __attribute__((ext_vector_type(16)));

// ---------------- GEMM tiling ----------------
#define BM 128
#define BN 64
#define KB 32          // one K=32 WMMA step per LDS slab
#define AS 40          // row stride in bf16 elements: 80 B, 16B-aligned, conflict-free

__device__ __forceinline__ void bf16_split(float v, __bf16& h, __bf16& l) {
    h = (__bf16)v;
    l = (__bf16)(v - (float)h);
}

// A-fragment (16x32 bf16): lanes 0-15 need K{ko..ko+7, ko+16..ko+23}, ko = 0;
// lanes 16-31 the same with ko = 8.
__device__ __forceinline__ v16bf frag_a(const __bf16* p, int row, int ko) {
    v8bf x0 = *(const v8bf*)(p + row * AS + ko);
    v8bf x1 = *(const v8bf*)(p + row * AS + ko + 16);
    return __builtin_shufflevector(x0, x1, 0,1,2,3,4,5,6,7,8,9,10,11,12,13,14,15);
}

// B-fragment (32x16 bf16): lanes 0-15 carry K{0..7,16..23}, lanes 16-31 carry K{8..15,24..31}.
__device__ __forceinline__ v16bf frag_b(const __bf16* p, int row, int ko) {
    v8bf x0 = *(const v8bf*)(p + row * AS + ko);
    v8bf x1 = *(const v8bf*)(p + row * AS + ko + 16);
    return __builtin_shufflevector(x0, x1, 0,1,2,3,4,5,6,7,8,9,10,11,12,13,14,15);
}

__global__ __launch_bounds__(256) void gate_gemm_bf16x2_wmma(
    const float* __restrict__ x,      // (N, DIM) row-major f32
    const float* __restrict__ w,      // (E, DIM) row-major f32
    float* __restrict__ logits)       // (N, E) row-major f32
{
    __shared__ __bf16 aHi[BM * AS];
    __shared__ __bf16 aLo[BM * AS];
    __shared__ __bf16 bHi[BN * AS];
    __shared__ __bf16 bLo[BN * AS];

    const int t       = threadIdx.x;
    const int wave    = t >> 5;
    const int lane    = t & 31;
    const int rowBase = blockIdx.x * BM;
    const int colBase = blockIdx.y * BN;

    const int frow = lane & 15;            // row within 16x16 subtile
    const int koA  = (lane >> 4) * 8;      // A fragment K offset: 0 or 8
    const int koB  = (lane >> 4) * 8;      // B fragment K offset: 0 or 8
    const int mrow = wave * 16 + frow;     // this wave's A row in the tile

    v8f acc[4];
    #pragma unroll
    for (int cb = 0; cb < 4; ++cb) acc[cb] = (v8f){0.f,0.f,0.f,0.f,0.f,0.f,0.f,0.f};

    // Cooperative slab loads (256 threads, KB=32 f32 per row):
    //   A: 128 rows -> thread t: 16 consecutive floats of row t/2
    //   B:  64 rows -> thread t:  8 consecutive floats of row t/4
    const int ar = t >> 1;
    const int ac = (t & 1) * 16;
    const int br = t >> 2;
    const int bc = (t & 3) * 8;

    const float* gA = x + (size_t)(rowBase + ar) * GATE_DIM + ac;
    const float* gB = w + (size_t)(colBase + br) * GATE_DIM + bc;

    for (int k0 = 0; k0 < GATE_DIM; k0 += KB) {
        __syncthreads();   // previous slab fully consumed
        {   // A slab: 16 floats -> 16 hi + 16 lo bf16
            const float4* g = (const float4*)(gA + k0);
            float4 f[4] = { g[0], g[1], g[2], g[3] };
            const float* fv = (const float*)f;
            v8bf h[2], l[2];
            #pragma unroll
            for (int i = 0; i < 16; ++i) {
                __bf16 hh, ll;
                bf16_split(fv[i], hh, ll);
                h[i >> 3][i & 7] = hh;
                l[i >> 3][i & 7] = ll;
            }
            *(v8bf*)&aHi[ar * AS + ac]     = h[0];
            *(v8bf*)&aHi[ar * AS + ac + 8] = h[1];
            *(v8bf*)&aLo[ar * AS + ac]     = l[0];
            *(v8bf*)&aLo[ar * AS + ac + 8] = l[1];
        }
        {   // B slab: 8 floats -> 8 hi + 8 lo bf16
            const float4* g = (const float4*)(gB + k0);
            float4 f[2] = { g[0], g[1] };
            const float* fv = (const float*)f;
            v8bf h, l;
            #pragma unroll
            for (int i = 0; i < 8; ++i) {
                __bf16 hh, ll;
                bf16_split(fv[i], hh, ll);
                h[i] = hh;
                l[i] = ll;
            }
            *(v8bf*)&bHi[br * AS + bc] = h;
            *(v8bf*)&bLo[br * AS + bc] = l;
        }
        __syncthreads();   // slab visible to all waves

        const v16bf ah = frag_a(aHi, mrow, koA);
        const v16bf al = frag_a(aLo, mrow, koA);
        #pragma unroll
        for (int cb = 0; cb < 4; ++cb) {
            const int nrow = cb * 16 + frow;
            v16bf bh = frag_b(bHi, nrow, koB);
            v16bf bl = frag_b(bLo, nrow, koB);
            acc[cb] = __builtin_amdgcn_wmma_f32_16x16x32_bf16(
                false, ah, false, bh, (short)0, acc[cb], false, false);
            acc[cb] = __builtin_amdgcn_wmma_f32_16x16x32_bf16(
                false, ah, false, bl, (short)0, acc[cb], false, false);
            acc[cb] = __builtin_amdgcn_wmma_f32_16x16x32_bf16(
                false, al, false, bh, (short)0, acc[cb], false, false);
        }
    }

    // C/D layout: VGPR i, lanes 0-15 -> M=i, N=lane; lanes 16-31 -> M=8+i, N=lane-16
    const int rofs = (lane >> 4) << 3;
    #pragma unroll
    for (int cb = 0; cb < 4; ++cb) {
        const int col = colBase + cb * 16 + frow;
        #pragma unroll
        for (int i = 0; i < 8; ++i) {
            const int row = rowBase + wave * 16 + i + rofs;
            logits[(size_t)row * GATE_E + col] = acc[cb][i];
        }
    }
}

// ---------------- Routing: one wave32 per token (8 experts per lane) ----------------
__global__ __launch_bounds__(256) void gate_route(
    const float* __restrict__ logits,  // (N, E)
    const float* __restrict__ bias,    // (E,)
    float* __restrict__ outW,          // (N, 8) fp32
    long long* __restrict__ outI)      // (N, 8) int64
{
    const int lane  = threadIdx.x & 31;
    const int wave  = threadIdx.x >> 5;
    const int token = blockIdx.x * 8 + wave;

    const float* lg = logits + (size_t)token * GATE_E + lane * 8;
    float raw[8];
    {
        float4 r0 = ((const float4*)lg)[0];
        float4 r1 = ((const float4*)lg)[1];
        raw[0]=r0.x; raw[1]=r0.y; raw[2]=r0.z; raw[3]=r0.w;
        raw[4]=r1.x; raw[5]=r1.y; raw[6]=r1.z; raw[7]=r1.w;
    }

    // --- softmax over 256 (fp32) ---
    float m = raw[0];
    #pragma unroll
    for (int j = 1; j < 8; ++j) m = fmaxf(m, raw[j]);
    #pragma unroll
    for (int off = 16; off > 0; off >>= 1) m = fmaxf(m, __shfl_xor(m, off, 32));

    float sc[8]; float sum = 0.f;
    #pragma unroll
    for (int j = 0; j < 8; ++j) { sc[j] = __expf(raw[j] - m); sum += sc[j]; }
    #pragma unroll
    for (int off = 16; off > 0; off >>= 1) sum += __shfl_xor(sum, off, 32);
    const float inv = 1.f / sum;

    const float* bp = bias + lane * 8;
    {
        float4 b0 = ((const float4*)bp)[0];
        float4 b1 = ((const float4*)bp)[1];
        sc[0]=sc[0]*inv+b0.x; sc[1]=sc[1]*inv+b0.y; sc[2]=sc[2]*inv+b0.z; sc[3]=sc[3]*inv+b0.w;
        sc[4]=sc[4]*inv+b1.x; sc[5]=sc[5]*inv+b1.y; sc[6]=sc[6]*inv+b1.z; sc[7]=sc[7]*inv+b1.w;
    }

    // --- group pruning: 8 groups of 32 experts = 4 lanes each ---
    float gmax = sc[0];
    #pragma unroll
    for (int j = 1; j < 8; ++j) gmax = fmaxf(gmax, sc[j]);
    gmax = fmaxf(gmax, __shfl_xor(gmax, 1, 32));
    gmax = fmaxf(gmax, __shfl_xor(gmax, 2, 32));   // all 4 lanes of group hold group max

    const int myg = lane >> 2;
    int rank = 0;
    #pragma unroll
    for (int g = 0; g < 8; ++g) {
        float gm = __shfl(gmax, g * 4, 32);
        rank += (gm > gmax) || ((gm == gmax) && (g < myg));
    }
    if (rank >= 4) {
        #pragma unroll
        for (int j = 0; j < 8; ++j) sc[j] = -INFINITY;
    }

    // --- iterative top-8 (tie-break: lowest index, matches lax.top_k) ---
    #pragma unroll
    for (int it = 0; it < GATE_TOPK; ++it) {
        float v = sc[0]; int idx = lane * 8;
        #pragma unroll
        for (int j = 1; j < 8; ++j)
            if (sc[j] > v) { v = sc[j]; idx = lane * 8 + j; }
        #pragma unroll
        for (int off = 16; off > 0; off >>= 1) {
            float ov = __shfl_xor(v, off, 32);
            int   oi = __shfl_xor(idx, off, 32);
            if (ov > v || (ov == v && oi < idx)) { v = ov; idx = oi; }
        }
        if ((idx >> 3) == lane) {               // owning lane writes result
            outW[(size_t)token * GATE_TOPK + it] = raw[idx & 7] * 2.5f;  // raw logit * route_scale
            outI[(size_t)token * GATE_TOPK + it] = (long long)idx;
            sc[idx & 7] = -INFINITY;
        }
    }
}

// ---------------------------------------------------------------------------
extern "C" void kernel_launch(void* const* d_in, const int* in_sizes, int n_in,
                              void* d_out, int out_size, void* d_ws, size_t ws_size,
                              hipStream_t stream) {
    (void)in_sizes; (void)n_in; (void)out_size; (void)ws_size;
    const float* x    = (const float*)d_in[0];   // (N, DIM) f32
    const float* w    = (const float*)d_in[1];   // (E, DIM) f32
    const float* bias = (const float*)d_in[2];   // (E,)    f32

    float* logits = (float*)d_ws;                                  // N*E*4 = 8 MB scratch
    float*     outW = (float*)d_out;                               // (N, 8) f32
    long long* outI = (long long*)((char*)d_out +
                       (size_t)GATE_N * GATE_TOPK * sizeof(float)); // (N, 8) i64

    dim3 g1(GATE_N / BM, GATE_E / BN, 1);        // 64 x 4
    gate_gemm_bf16x2_wmma<<<g1, 256, 0, stream>>>(x, w, logits);

    gate_route<<<GATE_N / 8, 256, 0, stream>>>(logits, bias, outW, outI);
}
